// Bidirectional_LSTM_Decoder_87995289960807
// MI455X (gfx1250) — compile-verified, matrix operand order checked by reading, not run on hardware
//
#include <hip/hip_runtime.h>

// Bidirectional LSTM decoder for MI455X (gfx1250, wave32, WMMA + TDM).
//
//  1. pack_x / pack_w: fp32 -> bf16 fragment blocks (ISA 7.12.2 16-bit layout,
//     per-lane contiguous 32B so fragment loads are 2x b128).
//  2. xgemm: Xg[d][s] = x[t(d,s)] @ Wih_d^T + b_d  (parallel WMMA GEMM, f32 acc).
//  3. recurrent: 2 dirs x 16 WGs. Each WG TDM-loads its 128KB Whh column slice
//     into LDS once (one tensor_load_to_lds), then runs 512 serial steps of
//     h @ Whh^T via v_wmma from LDS. The per-step gate preactivation slab
//     (32KB) is double-buffered in LDS and prefetched one step ahead with a 3D
//     TDM load overlapped with compute (s_wait_tensorcnt pipelining). c stays
//     in registers in C-fragment layout; h is exchanged through a
//     double-buffered global bf16 buffer with an atomic step barrier
//     (+ s_cluster_barrier, NOP when unclustered).
//
// Workspace: ~552.3 MB (xpk 32MB | wpk 8MB | Xg 512MB | hbuf 256KB | sync).

#define B_    64
#define T_    512
#define IN_   512
#define H_    512
#define FOURH 2048

typedef float        v8f   __attribute__((ext_vector_type(8)));
typedef __bf16       v16bf __attribute__((ext_vector_type(16)));
typedef unsigned int u32x4 __attribute__((ext_vector_type(4)));
typedef int          i32x8 __attribute__((ext_vector_type(8)));
typedef int          i32x4 __attribute__((ext_vector_type(4)));

union Frag { uint4 q[2]; unsigned int u[8]; v16bf v; };

// K offset pattern for 16-bit A/B fragments (ISA 7.12.2): vgpr r holds the bf16
// pair (K, K+1) with K = (r/4)*16 + (r%4)*2 + (lane/16)*8.
__device__ __forceinline__ int kpat(int lane, int r) {
  return ((r >> 2) << 4) | ((r & 3) << 1) | ((lane >> 4) << 3);
}

__device__ __forceinline__ unsigned short f2bf(float f) {
  unsigned int u = __float_as_uint(f);
  u += 0x7FFFu + ((u >> 16) & 1u);   // round-to-nearest-even
  return (unsigned short)(u >> 16);
}

__device__ __forceinline__ v8f wmma_bf16(v16bf a, v16bf b, v8f c) {
  return __builtin_amdgcn_wmma_f32_16x16x32_bf16(false, a, false, b, (short)0, c,
                                                 false, false);
}

__device__ __forceinline__ float sigmoidf_(float x) {
  return __fdividef(1.0f, 1.0f + __expf(-x));
}
__device__ __forceinline__ float tanhf_(float x) {
  float a = __expf(-2.0f * fabsf(x));
  float t = __fdividef(1.0f - a, 1.0f + a);
  return copysignf(t, x);
}

// ----------------------------------------------------------------- TDM helper
// Build a Tensor DMA Descriptor (CDNA5 ISA) and issue
// TENSOR_LOAD_TO_LDS (6-arg clang-23 builtin form). Up to 3D tile:
// x (contiguous, t0 units), y (stride s0), z (stride s1). All values are
// workgroup-uniform -> SGPR descriptor groups.
__device__ __forceinline__ void tdm_load(unsigned lds_off, unsigned long long ga,
                                         int dsz,  // 0:1B 1:2B 2:4B 3:8B
                                         unsigned t0, unsigned t1, unsigned t2,
                                         unsigned long long n0, unsigned n1,
                                         unsigned n2,
                                         unsigned long long s0,
                                         unsigned long long s1) {
  u32x4 g0;
  g0[0] = 1u;                                        // count=1, user descriptor
  g0[1] = lds_off;                                   // lds_addr (bytes)
  g0[2] = (unsigned)(ga & 0xFFFFFFFFull);            // global_addr[31:0]
  g0[3] = (unsigned)((ga >> 32) & 0x1FFFFFFull) | (2u << 30);  // [56:32] | type=2
  i32x8 g1;
  g1[0] = (dsz & 3) << 16;                           // wg_mask=0 | data_size
  g1[1] = (int)((n0 & 0xFFFFull) << 16);             // tensor_dim0[15:0]
  g1[2] = (int)(((n0 >> 16) & 0xFFFFull) | ((unsigned long long)(n1 & 0xFFFFu) << 16));
  g1[3] = (int)(((n1 >> 16) & 0xFFFFu) | (t0 << 16));          // | tile_dim0
  g1[4] = (int)((t1 & 0xFFFFu) | (t2 << 16));        // tile_dim1 | tile_dim2
  g1[5] = (int)(s0 & 0xFFFFFFFFull);                 // tensor_dim0_stride[31:0]
  g1[6] = (int)(((s0 >> 32) & 0xFFFFull) | ((s1 & 0xFFFFull) << 16));
  g1[7] = (int)((s1 >> 16) & 0xFFFFFFFFull);         // tensor_dim1_stride[47:16]
  i32x4 g2;
  g2[0] = (int)n2; g2[1] = 0; g2[2] = 0; g2[3] = 0;  // tensor_dim2
  i32x4 g3 = {0, 0, 0, 0};
  i32x8 g4 = {0, 0, 0, 0, 0, 0, 0, 0};
  __builtin_amdgcn_tensor_load_to_lds(g0, g1, g2, g3, g4, 0);
}

// Cross-workgroup step barrier: 16 WGs per direction share one monotonic counter.
__device__ __forceinline__ void gsync(unsigned int* cnt, int epoch) {
  __syncthreads();
  if (threadIdx.x == 0) {
    __threadfence();
    atomicAdd(cnt, 1u);
    unsigned int target = 16u * (unsigned int)epoch;
    while (*(volatile unsigned int*)cnt < target) __builtin_amdgcn_s_sleep(1);
  }
  __builtin_amdgcn_s_cluster_barrier();  // NOP when not launched as a cluster
  __syncthreads();
  __threadfence();
}

// ---------------------------------------------------------------- pack kernels
__global__ void pack_x_kernel(const float* __restrict__ x,
                              unsigned int* __restrict__ xpk) {
  unsigned int idx = blockIdx.x * 256u + threadIdx.x;   // < 8,388,608 dwords
  int r = idx & 7, lane = (idx >> 3) & 31, kt = (idx >> 8) & 15;
  int m = (idx >> 12) & 3, t = idx >> 14;
  int b  = m * 16 + (lane & 15);
  int k0 = kt * 32 + kpat(lane, r);
  const float2 p = *(const float2*)(x + ((size_t)b * T_ + t) * IN_ + k0);
  xpk[idx] = (unsigned int)f2bf(p.x) | ((unsigned int)f2bf(p.y) << 16);
}

__global__ void pack_w_kernel(const float* __restrict__ w0, const float* __restrict__ w1,
                              const float* __restrict__ w2, const float* __restrict__ w3,
                              unsigned int* __restrict__ wpk) {
  unsigned int idx = blockIdx.x * 256u + threadIdx.x;   // < 2,097,152 dwords
  int r = idx & 7, lane = (idx >> 3) & 31, kt = (idx >> 8) & 15;
  int nt = (idx >> 12) & 127, mat = idx >> 19;
  const float* w = (mat == 0) ? w0 : (mat == 1) ? w1 : (mat == 2) ? w2 : w3;
  int n  = nt * 16 + (lane & 15);
  int k0 = kt * 32 + kpat(lane, r);
  const float2 p = *(const float2*)(w + (size_t)n * IN_ + k0);
  wpk[idx] = (unsigned int)f2bf(p.x) | ((unsigned int)f2bf(p.y) << 16);
}

// ------------------------------------------------ input-projection GEMM (WMMA)
__global__ void xgemm_kernel(const unsigned int* __restrict__ xpk,
                             const unsigned int* __restrict__ wpk,
                             const float* __restrict__ bias_f,
                             const float* __restrict__ bias_b,
                             float* __restrict__ xg) {
  unsigned int wid = blockIdx.x * 8u + (threadIdx.x >> 5);  // < 524,288 waves
  int lane = threadIdx.x & 31;
  int nt = wid & 127, m = (wid >> 7) & 3, s = (wid >> 9) & 511, d = wid >> 18;
  int t = d ? (T_ - 1 - s) : s;  // backward direction consumes reversed time

  const float* bias = d ? bias_b : bias_f;
  const unsigned int* wih = wpk + (size_t)(d ? 2 : 0) * 524288u;
  float bv = bias[nt * 16 + (lane & 15)];
  v8f acc;
#pragma unroll
  for (int r = 0; r < 8; ++r) acc[r] = bv;

  const unsigned int* xa = xpk + ((size_t)t * 4 + m) * 4096u + (unsigned)lane * 8u;
  const unsigned int* wb = wih + (size_t)nt * 4096u + (unsigned)lane * 8u;
#pragma unroll 4
  for (int kt = 0; kt < 16; ++kt) {
    Frag a, b;
    a.q[0] = ((const uint4*)(xa + kt * 256))[0];
    a.q[1] = ((const uint4*)(xa + kt * 256))[1];
    b.q[0] = ((const uint4*)(wb + kt * 256))[0];
    b.q[1] = ((const uint4*)(wb + kt * 256))[1];
    acc = wmma_bf16(a.v, b.v, acc);
  }

  int hi = lane >> 4, ln = lane & 15;
  float* o = xg + ((size_t)d * T_ + s) * (B_ * FOURH)
                + (size_t)(m * 16 + hi * 8) * FOURH + nt * 16 + ln;
#pragma unroll
  for (int r = 0; r < 8; ++r) o[(size_t)r * FOURH] = acc[r];
}

// -------------------------------------------------------- recurrent scan (WMMA)
// LDS map (dwords): [0,32768)        Whh slice  [8 lnt][16 kt][32 lane][8]
//                   [32768,49152)    Xg double buffer: 2 x [4 gi][64 row][32]
__global__ void recurrent_kernel(const float* __restrict__ xg,
                                 const unsigned int* __restrict__ wpk,
                                 const float* __restrict__ enc_h,
                                 const float* __restrict__ enc_c,
                                 unsigned short* __restrict__ hbuf,
                                 unsigned int* __restrict__ syncc,
                                 float* __restrict__ out) {
  extern __shared__ unsigned int ldsW[];
  const int d    = blockIdx.x >> 4;
  const int wg   = blockIdx.x & 15;
  const int wave = threadIdx.x >> 5;
  const int lane = threadIdx.x & 31;
  const int hi = lane >> 4, ln = lane & 15;
  const int m  = wave >> 1;         // batch tile 0..3
  const int nn = wave & 1;          // h-column sub-tile
  const int n0h = (wg * 2 + nn) * 16;

  const unsigned int* whh = wpk + (size_t)(d ? 3 : 1) * 524288u;
  const float* xgd = xg + (size_t)d * T_ * B_ * FOURH;
  unsigned short* hb0 = hbuf + (size_t)(d * 2 + 0) * (B_ * H_);
  unsigned short* hb1 = hbuf + (size_t)(d * 2 + 1) * (B_ * H_);
  unsigned int* cnt = syncc + d;

  if (wave == 0) {
    // One 2D TDM load stages the whole 128KB Whh slice: 4 rows (gates) of
    // 32KB (this WG's two contiguous column-tile blocks), gate stride 512KB.
    tdm_load(0, (unsigned long long)(uintptr_t)(whh + (size_t)wg * 8192u),
             /*dsz=*/3, /*tile*/ 4096, 4, 0, /*tensor*/ 4096ull, 4, 0,
             /*s0=*/65536ull, /*s1=*/0ull);
    // 3D TDM load of step-0 gate preactivations: x=32 cols, y=64 rows
    // (stride 2048), z=4 gates (stride 512) -> compacted LDS slab.
    tdm_load(131072u, (unsigned long long)(uintptr_t)(xgd + (size_t)wg * 32u),
             /*dsz=*/2, 32, 64, 4, 32ull, 64, 4, 2048ull, 512ull);
  }

  // h0 into exchange buffer (bf16), c0 into registers (C-fragment layout).
  for (int e = threadIdx.x; e < 2048; e += 256) {
    int row = e >> 5;
    int col = wg * 32 + (e & 31);
    hb0[row * H_ + col] = f2bf(enc_h[(size_t)row * (2 * H_) + d * H_ + col]);
  }
  float c[8], hlast[8];
#pragma unroll
  for (int r = 0; r < 8; ++r) {
    c[r] = enc_c[(size_t)(m * 16 + hi * 8 + r) * (2 * H_) + d * H_ + n0h + ln];
    hlast[r] = 0.0f;
  }
  gsync(cnt, 1);

  for (int s = 0; s < T_; ++s) {
    const int p = s & 1;
    const unsigned short* hrd = p ? hb1 : hb0;
    unsigned short* hwr       = p ? hb0 : hb1;

    // Prefetch next step's Xg slab into the other LDS buffer, then wait for
    // this step's slab (<=1 outstanding tensor op => current slab complete).
    if (wave == 0) {
      if (s + 1 < T_) {
        tdm_load(131072u + (unsigned)((s + 1) & 1) * 32768u,
                 (unsigned long long)(uintptr_t)
                     (xgd + (size_t)(s + 1) * (B_ * FOURH) + (size_t)wg * 32u),
                 2, 32, 64, 4, 32ull, 64, 4, 2048ull, 512ull);
        __builtin_amdgcn_s_wait_tensorcnt(1);
      } else {
        __builtin_amdgcn_s_wait_tensorcnt(0);
      }
    }
    __syncthreads();

    // C init = precomputed x@Wih^T + b, read from the LDS slab.
    const unsigned int* xls = ldsW + 32768u + (unsigned)p * 8192u;
    v8f acc[4];
#pragma unroll
    for (int gi = 0; gi < 4; ++gi)
#pragma unroll
      for (int r = 0; r < 8; ++r)
        acc[gi][r] = __uint_as_float(
            xls[gi * 2048 + (m * 16 + hi * 8 + r) * 32 + nn * 16 + ln]);

    // K loop: A fragment from previous h (global bf16), B fragments from LDS.
    const unsigned short* hrow = hrd + (m * 16 + ln) * H_;
    for (int kt = 0; kt < 16; ++kt) {
      Frag a;
#pragma unroll
      for (int r = 0; r < 8; ++r)
        a.u[r] = *(const unsigned int*)(hrow + kt * 32 + kpat(lane, r));
#pragma unroll
      for (int gi = 0; gi < 4; ++gi) {
        Frag b;
        const uint4* lsrc =
            (const uint4*)(ldsW + (((gi * 2 + nn) * 16 + kt) * 32 + lane) * 8);
        b.q[0] = lsrc[0];
        b.q[1] = lsrc[1];
        acc[gi] = wmma_bf16(a.v, b.v, acc[gi]);
      }
    }

    const int tout = d ? (T_ - 1 - s) : s;
#pragma unroll
    for (int r = 0; r < 8; ++r) {
      float iv = sigmoidf_(acc[0][r]);
      float fv = sigmoidf_(acc[1][r]);
      float gv = tanhf_(acc[2][r]);
      float ov = sigmoidf_(acc[3][r]);
      c[r] = fv * c[r] + iv * gv;
      float hv = ov * tanhf_(c[r]);
      hlast[r] = hv;
      int brow = m * 16 + hi * 8 + r;
      int col  = n0h + ln;
      out[((size_t)brow * T_ + tout) * (2 * H_) + d * H_ + col] = hv;
      hwr[brow * H_ + col] = f2bf(hv);
    }
    __threadfence();
    gsync(cnt, s + 2);
  }

  // Final h and c (concatenated after `out` in d_out).
  float* outH = out + (size_t)B_ * T_ * (2 * H_);
  float* outC = outH + (size_t)B_ * (2 * H_);
#pragma unroll
  for (int r = 0; r < 8; ++r) {
    int brow = m * 16 + hi * 8 + r;
    int col  = d * H_ + n0h + ln;
    outH[(size_t)brow * (2 * H_) + col] = hlast[r];
    outC[(size_t)brow * (2 * H_) + col] = c[r];
  }
}

__global__ void zero_sync_kernel(unsigned int* syncc) {
  if (threadIdx.x < 8) syncc[threadIdx.x] = 0u;
}

// ----------------------------------------------------------------------- host
extern "C" void kernel_launch(void* const* d_in, const int* in_sizes, int n_in,
                              void* d_out, int out_size, void* d_ws, size_t ws_size,
                              hipStream_t stream) {
  (void)in_sizes; (void)n_in; (void)out_size; (void)ws_size;
  const float* x     = (const float*)d_in[0];
  const float* enc_h = (const float*)d_in[1];
  const float* enc_c = (const float*)d_in[2];
  const float* Wih_f = (const float*)d_in[3];
  const float* Whh_f = (const float*)d_in[4];
  const float* b_f   = (const float*)d_in[5];
  const float* Wih_b = (const float*)d_in[6];
  const float* Whh_b = (const float*)d_in[7];
  const float* b_b   = (const float*)d_in[8];
  float* out = (float*)d_out;

  char* ws = (char*)d_ws;
  size_t off = 0;
  unsigned int*   xpk = (unsigned int*)(ws + off);   off += (size_t)32  << 20;
  unsigned int*   wpk = (unsigned int*)(ws + off);   off += (size_t)8   << 20;
  float*          xg  = (float*)(ws + off);          off += (size_t)512 << 20;
  unsigned short* hb  = (unsigned short*)(ws + off); off += (size_t)256 << 10;
  unsigned int*   sc  = (unsigned int*)(ws + off);

  zero_sync_kernel<<<1, 64, 0, stream>>>(sc);
  pack_x_kernel<<<32768, 256, 0, stream>>>(x, xpk);
  pack_w_kernel<<<8192, 256, 0, stream>>>(Wih_f, Whh_f, Wih_b, Whh_b, wpk);
  xgemm_kernel<<<65536, 256, 0, stream>>>(xpk, wpk, b_f, b_b, xg);
  recurrent_kernel<<<32, 256, 196608, stream>>>(xg, wpk, enc_h, enc_c, hb, sc, out);
}
